// gnet_62199716381113
// MI455X (gfx1250) — matrix operand order repaired and checked by reading, not compiled, not run on hardware
//
#include <hip/hip_runtime.h>
#include <hip/hip_bf16.h>

// Problem constants (match reference)
#define GN 20000
#define GE 160000
#define GBATCH 16
#define DIN 512
#define DMID 512
#define DOUT 384

typedef __attribute__((ext_vector_type(16))) __bf16 v16bf;
typedef __attribute__((ext_vector_type(4)))  __bf16 v4bf;
typedef __attribute__((ext_vector_type(8)))  float  v8f;

// ---------------------------------------------------------------------------
// WMMA GEMM: OUT[M,Nc] (+)= A[M,K] @ W[K,Nc]   (bf16 multiply, f32 accumulate)
// Block: 256 threads (8 wave32) -> 128x64 output tile, each wave 16x64.
// Double-buffered LDS; branch-free vectorized staging (b128 loads, b64 stores).
// ---------------------------------------------------------------------------
#define TM 128
#define TN 64
#define TK 32
#define PAD 8

__global__ __launch_bounds__(256) void gemm_bf16_wmma(
    const float* __restrict__ A, const float* __restrict__ W,
    const float* __restrict__ bias, float* __restrict__ OUT,
    int M, int K, int Nc, int accumulate, int relu)
{
    __shared__ __bf16 sA[2][TM][TK + PAD];   // [buf][row][k]
    __shared__ __bf16 sB[2][TN][TK + PAD];   // [buf][col][k] (transposed W tile)

    const int tid  = threadIdx.x;
    const int lane = tid & 31;
    const int wave = tid >> 5;
    const int m0 = blockIdx.x * TM;
    const int n0 = blockIdx.y * TN;

    // Per-thread staging coordinates (A: 4 x float4, W: 2 x float4)
    int aR[4], aC[4], aOK[4];
#pragma unroll
    for (int i = 0; i < 4; ++i) {
        int idx = i * 256 + tid;          // 0..1023 over 128x8 float4 slots
        aR[i] = idx >> 3;                 // row in tile
        aC[i] = (idx & 7) << 2;           // k offset (float index)
        int grow = m0 + aR[i];
        aOK[i] = (grow < M);
    }
    int wR[2], wC[2];
#pragma unroll
    for (int i = 0; i < 2; ++i) {
        int idx = i * 256 + tid;          // 0..511 over 32x16 float4 slots
        wR[i] = idx >> 4;                 // k in tile
        wC[i] = (idx & 15) << 2;          // col offset
    }

    const int kSteps = K / TK;
    float4 aReg[4], wReg[2];

    // fetch tile `ks` into registers (branch-free: clamp row, mask later)
    auto fetch = [&](int ks) {
        const int k0 = ks * TK;
#pragma unroll
        for (int i = 0; i < 4; ++i) {
            int grow = m0 + aR[i];
            int crow = aOK[i] ? grow : (M - 1);
            aReg[i] = *(const float4*)(A + (long)crow * K + (k0 + aC[i]));
        }
#pragma unroll
        for (int i = 0; i < 2; ++i) {
            wReg[i] = *(const float4*)(W + (long)(k0 + wR[i]) * Nc + (n0 + wC[i]));
        }
    };
    // convert + store registers into LDS buffer b
    auto stash = [&](int b) {
#pragma unroll
        for (int i = 0; i < 4; ++i) {
            float x0 = aOK[i] ? aReg[i].x : 0.0f;
            float x1 = aOK[i] ? aReg[i].y : 0.0f;
            float x2 = aOK[i] ? aReg[i].z : 0.0f;
            float x3 = aOK[i] ? aReg[i].w : 0.0f;
            v4bf pk;
            pk[0] = (__bf16)x0; pk[1] = (__bf16)x1;
            pk[2] = (__bf16)x2; pk[3] = (__bf16)x3;
            *(v4bf*)&sA[b][aR[i]][aC[i]] = pk;          // ds_store_b64
        }
#pragma unroll
        for (int i = 0; i < 2; ++i) {                    // transposed scatter
            sB[b][wC[i] + 0][wR[i]] = (__bf16)wReg[i].x;
            sB[b][wC[i] + 1][wR[i]] = (__bf16)wReg[i].y;
            sB[b][wC[i] + 2][wR[i]] = (__bf16)wReg[i].z;
            sB[b][wC[i] + 3][wR[i]] = (__bf16)wReg[i].w;
        }
    };

    v8f zero8 = {0.f,0.f,0.f,0.f,0.f,0.f,0.f,0.f};
    v8f acc[4] = {zero8, zero8, zero8, zero8};

    // Prologue: stage tile 0 into buffer 0
    fetch(0);
    stash(0);

    const int arow = wave * 16 + (lane & 15);
    const int ah   = (lane < 16) ? 0 : 8;
    const int bk   = (lane < 16) ? 0 : 8;

    for (int ks = 0; ks < kSteps; ++ks) {
        __syncthreads();
        const int cur = ks & 1;
        const bool more = (ks + 1) < kSteps;
        if (more) fetch(ks + 1);          // overlap next-tile loads with WMMA

        // A fragment (16x32): lanes 0-15 K 0..7 & 16..23, lanes 16-31 K 8..15 & 24..31
        v16bf afrag;
#pragma unroll
        for (int i = 0; i < 8; ++i) afrag[i]     = sA[cur][arow][ah + i];
#pragma unroll
        for (int i = 0; i < 8; ++i) afrag[8 + i] = sA[cur][arow][16 + ah + i];

#pragma unroll
        for (int cb = 0; cb < 4; ++cb) {
            const int bcol = cb * 16 + (lane & 15);
            v16bf bfrag;
#pragma unroll
            for (int i = 0; i < 16; ++i) bfrag[i] = sB[cur][bcol][(i < 8) ? (bk + i) : (16 + bk + i - 8)];
            acc[cb] = __builtin_amdgcn_wmma_f32_16x16x32_bf16(
                false, afrag, false, bfrag, (short)0, acc[cb], false, false);
        }
        if (more) stash((ks + 1) & 1);    // write other buffer; next sync publishes
    }

    // Write back: lanes 0-15 hold rows 0..7 (c[r]=M=r), lanes 16-31 rows 8..15
    const int mb   = (lane < 16) ? 0 : 8;
    const int ncol = lane & 15;
#pragma unroll
    for (int cb = 0; cb < 4; ++cb) {
        int gcol = n0 + cb * 16 + ncol;
#pragma unroll
        for (int r = 0; r < 8; ++r) {
            int grow = m0 + wave * 16 + mb + r;
            if (grow < M) {
                long o = (long)grow * Nc + gcol;
                float prev = accumulate ? OUT[o] : (bias ? bias[gcol] : 0.0f);
                float v = prev + acc[cb][r];
                if (relu) v = fmaxf(v, 0.0f);
                OUT[o] = v;
            }
        }
    }
}

// ---------------------------------------------------------------------------
// Small helper kernels
// ---------------------------------------------------------------------------
__global__ void zero_f32(float* p, long n) {
    long i = (long)blockIdx.x * blockDim.x + threadIdx.x;
    if (i < n) p[i] = 0.0f;
}

__global__ void deg_accum(const int* __restrict__ row, const float* __restrict__ w,
                          float* __restrict__ deg, int e) {
    int i = blockIdx.x * blockDim.x + threadIdx.x;
    if (i < e) atomicAdd(&deg[row[i]], w[i]);
}

__global__ void norm_kernel(const int* __restrict__ row, const int* __restrict__ col,
                            const float* __restrict__ w, const float* __restrict__ deg,
                            float* __restrict__ nrm, int e) {
    int i = blockIdx.x * blockDim.x + threadIdx.x;
    if (i < e) {
        float dr = deg[row[i]], dc = deg[col[i]];
        float ir = dr > 0.0f ? rsqrtf(dr) : 0.0f;
        float ic = dc > 0.0f ? rsqrtf(dc) : 0.0f;
        nrm[i] = -ir * w[i] * ic;
    }
}

// y = s*src (src==nullptr -> 0)
__global__ void init_scaled(float* __restrict__ y, const float* __restrict__ src,
                            float s, long n) {
    long i = (long)blockIdx.x * blockDim.x + threadIdx.x;
    if (i < n) y[i] = src ? s * src[i] : 0.0f;
}

// y[col[e]] += scale*norm[e]*h[row[e]]   (one thread per (edge, 4 features))
__global__ void prop_scatter(const int* __restrict__ row, const int* __restrict__ col,
                             const float* __restrict__ nrm, const float* __restrict__ h,
                             float* __restrict__ y, int e, int D, float scale) {
    long idx = (long)blockIdx.x * blockDim.x + threadIdx.x;
    int fpere = D >> 2;
    long total = (long)e * fpere;
    if (idx >= total) return;
    int ed = (int)(idx / fpere);
    int f  = ((int)(idx % fpere)) << 2;
    float w = nrm[ed] * scale;
    int r = row[ed], c = col[ed];
    const float4 hv = *(const float4*)(h + (long)r * D + f);
    float* yp = y + (long)c * D + f;
    atomicAdd(yp + 0, w * hv.x);
    atomicAdd(yp + 1, w * hv.y);
    atomicAdd(yp + 2, w * hv.z);
    atomicAdd(yp + 3, w * hv.w);
}

// gate[n] = tanh(dot(h2a[n,0:64], Wa3) + ba3)   (one wave32 per node)
__global__ __launch_bounds__(256) void gate_kernel(
    const float* __restrict__ ha2, const float* __restrict__ Wa3,
    const float* __restrict__ ba3, float* __restrict__ gate, int n) {
    int gwave = (int)(((long)blockIdx.x * blockDim.x + threadIdx.x) >> 5);
    int lane  = threadIdx.x & 31;
    if (gwave >= n) return;
    const float* hp = ha2 + (long)gwave * 64;
    float v = hp[lane] * Wa3[lane] + hp[lane + 32] * Wa3[lane + 32];
#pragma unroll
    for (int off = 16; off > 0; off >>= 1) v += __shfl_xor(v, off, 32);
    if (lane == 0) gate[gwave] = tanhf(v + ba3[0]);
}

__device__ __forceinline__ unsigned fkey(float f) {
    unsigned b = __float_as_uint(f);
    return (b & 0x80000000u) ? ~b : (b | 0x80000000u);
}
__device__ __forceinline__ float fdecode(unsigned k) {
    unsigned b = (k & 0x80000000u) ? (k & 0x7fffffffu) : ~k;
    return __uint_as_float(b);
}

__global__ void gmax_atomic(const float* __restrict__ gate, const int* __restrict__ batch,
                            unsigned* __restrict__ bits, int n) {
    int i = blockIdx.x * blockDim.x + threadIdx.x;
    if (i < n) atomicMax(&bits[batch[i]], fkey(gate[i]));
}

__global__ void gmax_decode(const unsigned* __restrict__ bits, float* __restrict__ gmaxf) {
    int i = threadIdx.x;
    if (i < GBATCH) gmaxf[i] = (bits[i] == 0u) ? 0.0f : fdecode(bits[i]);
}

__global__ void esum_kernel(const float* __restrict__ gate, const int* __restrict__ batch,
                            const float* __restrict__ gmaxf, float* __restrict__ e,
                            float* __restrict__ denom, int n) {
    int i = blockIdx.x * blockDim.x + threadIdx.x;
    if (i < n) {
        int b = batch[i];
        float ev = expf(gate[i] - gmaxf[b]);
        e[i] = ev;
        atomicAdd(&denom[b], ev);
    }
}

__global__ void pool_kernel(const float* __restrict__ h2, const float* __restrict__ e,
                            const float* __restrict__ denom, const int* __restrict__ batch,
                            float* __restrict__ out, int n, int d) {
    long idx = (long)blockIdx.x * blockDim.x + threadIdx.x;
    int fpere = d >> 2;
    long total = (long)n * fpere;
    if (idx >= total) return;
    int node = (int)(idx / fpere);
    int f = ((int)(idx % fpere)) << 2;
    int b = batch[node];
    float a = e[node] / (denom[b] + 1e-16f);
    const float4 hv = *(const float4*)(h2 + (long)node * d + f);
    float* op = out + (long)b * d + f;
    atomicAdd(op + 0, a * hv.x);
    atomicAdd(op + 1, a * hv.y);
    atomicAdd(op + 2, a * hv.z);
    atomicAdd(op + 3, a * hv.w);
}

// ---------------------------------------------------------------------------
// Host-side helpers
// ---------------------------------------------------------------------------
static void launch_gemm(const float* A, const float* W, const float* bias, float* OUT,
                        int M, int K, int Nc, int acc, int relu, hipStream_t s) {
    dim3 grid((unsigned)((M + TM - 1) / TM), (unsigned)(Nc / TN));
    gemm_bf16_wmma<<<grid, 256, 0, s>>>(A, W, bias, OUT, M, K, Nc, acc, relu);
}

static void launch_prop(const float* h, float* y, const float* initsrc, float initscale,
                        float scatscale, const int* row, const int* col, const float* nrm,
                        int D, hipStream_t s) {
    long nd = (long)GN * D;
    init_scaled<<<(unsigned)((nd + 255) / 256), 256, 0, s>>>(y, initsrc, initscale, nd);
    long total = (long)GE * (D / 4);
    prop_scatter<<<(unsigned)((total + 255) / 256), 256, 0, s>>>(row, col, nrm, h, y, GE, D, scatscale);
}

// ChebConv layer (K=5): OUT = relu(sum_k Tk @ Wc[k] + bias)
static void cheb_layer(const float* X, const float* Wc, const float* bias, float* OUT,
                       int Din, int Dout, const int* row, const int* col, const float* nrm,
                       float* bufA, float* bufB, float* bufC, hipStream_t s) {
    long wsz = (long)Din * Dout;
    // k = 0 : T0 = X
    launch_gemm(X, Wc, bias, OUT, GN, Din, Dout, /*acc=*/0, /*relu=*/0, s);
    // k = 1 : T1 = prop(T0)
    launch_prop(X, bufA, nullptr, 0.0f, 1.0f, row, col, nrm, Din, s);
    launch_gemm(bufA, Wc + 1 * wsz, nullptr, OUT, GN, Din, Dout, 1, 0, s);
    // k = 2 : T2 = 2*prop(T1) - T0
    launch_prop(bufA, bufB, X, -1.0f, 2.0f, row, col, nrm, Din, s);
    launch_gemm(bufB, Wc + 2 * wsz, nullptr, OUT, GN, Din, Dout, 1, 0, s);
    // k = 3 : T3 = 2*prop(T2) - T1
    launch_prop(bufB, bufC, bufA, -1.0f, 2.0f, row, col, nrm, Din, s);
    launch_gemm(bufC, Wc + 3 * wsz, nullptr, OUT, GN, Din, Dout, 1, 0, s);
    // k = 4 : T4 = 2*prop(T3) - T2   (bufA is dead, reuse)
    launch_prop(bufC, bufA, bufB, -1.0f, 2.0f, row, col, nrm, Din, s);
    launch_gemm(bufA, Wc + 4 * wsz, nullptr, OUT, GN, Din, Dout, 1, /*relu=*/1, s);
}

// ---------------------------------------------------------------------------
extern "C" void kernel_launch(void* const* d_in, const int* in_sizes, int n_in,
                              void* d_out, int out_size, void* d_ws, size_t ws_size,
                              hipStream_t stream) {
    (void)in_sizes; (void)n_in; (void)out_size; (void)ws_size;

    const float* x    = (const float*)d_in[0];
    const int*   eidx = (const int*)d_in[1];
    const float* ew   = (const float*)d_in[2];
    const int*   batch= (const int*)d_in[3];
    const float* W1c  = (const float*)d_in[4];
    const float* b1c  = (const float*)d_in[5];
    const float* W2c  = (const float*)d_in[6];
    const float* b2c  = (const float*)d_in[7];
    const float* Wa1  = (const float*)d_in[8];
    const float* ba1  = (const float*)d_in[9];
    const float* Wa2  = (const float*)d_in[10];
    const float* ba2  = (const float*)d_in[11];
    const float* Wa3  = (const float*)d_in[12];
    const float* ba3  = (const float*)d_in[13];
    const int* row = eidx;       // edge_index[0]
    const int* col = eidx + GE;  // edge_index[1]
    float* out = (float*)d_out;

    // Workspace carving (256B aligned); total ~196 MB
    char* p = (char*)d_ws;
    auto carve = [&](size_t bytes) -> char* {
        char* r = p; p += (bytes + 255) & ~(size_t)255; return r;
    };
    float* deg   = (float*)carve((size_t)GN * 4);
    float* nrm   = (float*)carve((size_t)GE * 4);
    float* bufA  = (float*)carve((size_t)GN * 512 * 4);
    float* bufB  = (float*)carve((size_t)GN * 512 * 4);
    float* bufC  = (float*)carve((size_t)GN * 512 * 4);
    float* h1    = (float*)carve((size_t)GN * 512 * 4);
    float* h2    = (float*)carve((size_t)GN * DOUT * 4);
    float* small = (float*)carve((size_t)3 * GBATCH * 4);
    unsigned* gbits = (unsigned*)small;
    float* gmaxf = small + GBATCH;
    float* denom = small + 2 * GBATCH;

    // --- Chebyshev normalization: norm = -d^-1/2 * w * d^-1/2 ---
    zero_f32<<<(GN + 255) / 256, 256, 0, stream>>>(deg, GN);
    deg_accum<<<(GE + 255) / 256, 256, 0, stream>>>(row, ew, deg, GE);
    norm_kernel<<<(GE + 255) / 256, 256, 0, stream>>>(row, col, ew, deg, nrm, GE);

    // --- Two ChebConv layers (WMMA GEMMs + atomic scatter props) ---
    cheb_layer(x,  W1c, b1c, h1, DIN,  DMID, row, col, nrm, bufA, bufB, bufC, stream);
    cheb_layer(h1, W2c, b2c, h2, DMID, DOUT, row, col, nrm, bufA, bufB, bufC, stream);

    // --- Global attention pooling (bufA/B/C free now) ---
    float* ha1  = bufA;          // [N,128]
    float* ha2  = bufB;          // [N,64]
    float* gate = bufC;          // [N]
    float* ebuf = bufC + GN;     // [N]
    launch_gemm(h2,  Wa1, ba1, ha1, GN, DOUT, 128, 0, /*relu=*/1, stream);
    launch_gemm(ha1, Wa2, ba2, ha2, GN, 128,  64,  0, /*relu=*/1, stream);
    gate_kernel<<<(GN * 32 + 255) / 256, 256, 0, stream>>>(ha2, Wa3, ba3, gate, GN);

    zero_f32<<<1, 64, 0, stream>>>(small, 3 * GBATCH);
    gmax_atomic<<<(GN + 255) / 256, 256, 0, stream>>>(gate, batch, gbits, GN);
    gmax_decode<<<1, GBATCH, 0, stream>>>(gbits, gmaxf);
    esum_kernel<<<(GN + 255) / 256, 256, 0, stream>>>(gate, batch, gmaxf, ebuf, denom, GN);

    zero_f32<<<(GBATCH * DOUT + 255) / 256, 256, 0, stream>>>(out, (long)GBATCH * DOUT);
    long ptotal = (long)GN * (DOUT / 4);
    pool_kernel<<<(unsigned)((ptotal + 255) / 256), 256, 0, stream>>>(
        h2, ebuf, denom, batch, out, GN, DOUT);
}
